// nce_loss_66606352827120
// MI455X (gfx1250) — matrix operand order repaired and checked by reading, not compiled, not run on hardware
//
#include <hip/hip_runtime.h>
#include <hip/hip_bf16.h>

#define EMBED  256
#define BATCH  16384
#define NSAMP  4096
#define VOCABN 50000

typedef __bf16 bf16x16 __attribute__((ext_vector_type(16)));
typedef __bf16 bf16x8  __attribute__((ext_vector_type(8)));
typedef float  f32x8   __attribute__((ext_vector_type(8)));

// Fast softplus for the GEMM epilogue: log(1+e^x) = max(x,0) + log(1 + e^-|x|).
// Argument of the log is in (1,2] -> v_log_f32 is accurate there; dropped
// sub-1e-7 terms are negligible in the summed loss. exp/log are TRANS ops
// that co-execute with WMMA.
__device__ __forceinline__ float softplus_fast(float x) {
    return fmaxf(x, 0.0f) + __logf(1.0f + __expf(-fabsf(x)));
}

// Precise softplus for the (tiny) true-logit kernel.
__device__ __forceinline__ float softplus_f(float x) {
    return fmaxf(x, 0.0f) + log1pf(expf(-fabsf(x)));
}

__device__ __forceinline__ float neg_log_expected_count(int id) {
    float idf = (float)id;
    float p   = (logf(idf + 2.0f) - logf(idf + 1.0f)) * (1.0f / logf((float)(VOCABN + 1)));
    float ec  = -expm1f((float)NSAMP * log1pf(-p));
    return -logf(ec);
}

// ---------------------------------------------------------------------------
// Kernel 1a: split embedding f32 -> planar bf16 (hi, lo). grid*256 == BATCH*EMBED.
// ---------------------------------------------------------------------------
__global__ __launch_bounds__(256)
void nce_emb_split(const float* __restrict__ emb,
                   __bf16* __restrict__ eh, __bf16* __restrict__ el) {
    size_t g = (size_t)blockIdx.x * 256 + threadIdx.x;
    float x  = emb[g];
    __bf16 h = (__bf16)x;
    eh[g] = h;
    el[g] = (__bf16)(x - (float)h);
}

// ---------------------------------------------------------------------------
// Kernel 1b: gather sampled rows of nce_weight, split f32 -> bf16 (hi, lo),
// and fold bias - log(expected_count) into a per-sample scalar.
// ---------------------------------------------------------------------------
__global__ __launch_bounds__(256)
void nce_prep(const int* __restrict__ sids, const float* __restrict__ W,
              const float* __restrict__ bias,
              __bf16* __restrict__ wh, __bf16* __restrict__ wl,
              float* __restrict__ sbias) {
    int s  = blockIdx.x;
    int t  = threadIdx.x;
    int id = sids[s];
    float x   = W[(size_t)id * EMBED + t];
    __bf16 h  = (__bf16)x;
    wh[(size_t)s * EMBED + t] = h;
    wl[(size_t)s * EMBED + t] = (__bf16)(x - (float)h);
    if (t == 0) sbias[s] = bias[id] + neg_log_expected_count(id);
}

// ---------------------------------------------------------------------------
// Kernel 2: fused GEMM (bf16x3 WMMA) + softplus + row reduction.
// grid = (8, BATCH/64); block = 256 threads = 8 waves.
// Block stages a 64-row A macro-tile in LDS (bf16 hi/lo, padded stride).
// Each wave: n-tiles nt = bx*8 + wave, stepping 64 (4 tiles); per n-tile its
// B fragment feeds 4 M-tiles (B L2 traffic /4). kc loop kept ROLLED so the
// A fragments are re-read from LDS each iteration (no cross-nt hoist/spill),
// and the next n-tile's B rows are prefetched during current-tile compute.
// ---------------------------------------------------------------------------
__global__ __launch_bounds__(256)
void nce_gemm(const __bf16* __restrict__ eh, const __bf16* __restrict__ el,
              const __bf16* __restrict__ wh, const __bf16* __restrict__ wl,
              const float* __restrict__ sbias, float* __restrict__ row_sum) {
    // 264-element row stride (528 B) => consecutive rows hit distinct banks.
    __shared__ alignas(16) __bf16 Ah[64][264];
    __shared__ alignas(16) __bf16 Al[64][264];

    const int t     = threadIdx.x;
    const int mbase = blockIdx.y * 64;

    // Cooperative fill: pure bf16 copy of the pre-split planes (16B chunks).
    {
        const bf16x8* src_h = (const bf16x8*)(eh + (size_t)mbase * EMBED);
        const bf16x8* src_l = (const bf16x8*)(el + (size_t)mbase * EMBED);
#pragma unroll
        for (int i = 0; i < 8; ++i) {
            int idx = i * 256 + t;      // 0..2047 chunks of 8 bf16
            int row = idx >> 5;
            int c8  = idx & 31;
            *(bf16x8*)&Ah[row][c8 * 8] = src_h[idx];
            *(bf16x8*)&Al[row][c8 * 8] = src_l[idx];
        }
    }
    __syncthreads();

    const int lane = t & 31;
    const int wave = t >> 5;
    const int mrow = lane & 15;   // A row within tile held by this lane
    const int half = lane >> 4;   // K-half select (WMMA lane layout)
    const int nb   = lane & 15;   // B column within tile

#pragma unroll 1
    for (int nt = blockIdx.x * 8 + wave; nt < NSAMP / 16; nt += 64) {
        const int n = nt * 16 + nb;
        const __bf16* bh_ptr = wh + (size_t)n * EMBED + half * 16;
        const __bf16* bl_ptr = wl + (size_t)n * EMBED + half * 16;

        // Prefetch next n-tile's B rows (512 B each, hi+lo) into cache.
        if (nt + 64 < NSAMP / 16) {   // wave-uniform: EXEC stays all-ones
            const char* ph = (const char*)(wh + (size_t)(n + 64 * 16) * EMBED);
            const char* pl = (const char*)(wl + (size_t)(n + 64 * 16) * EMBED);
#pragma unroll
            for (int c = 0; c < 512; c += 128) {
                __builtin_prefetch(ph + c, 0, 3);
                __builtin_prefetch(pl + c, 0, 3);
            }
        }

        int zofs = 0;
        asm volatile("" : "+v"(zofs));   // opaque 0: pin LDS reads inside nt loop

        f32x8 acc[4];
#pragma unroll
        for (int mt = 0; mt < 4; ++mt) acc[mt] = (f32x8){};

#pragma unroll 1
        for (int kc = 0; kc < 8; ++kc) {
            bf16x16 bh = *(const bf16x16*)(bh_ptr + kc * 32);
            bf16x16 bl = *(const bf16x16*)(bl_ptr + kc * 32);

            const int ka = kc * 32 + half * 16 + zofs;
#pragma unroll
            for (int mt = 0; mt < 4; ++mt) {
                const int r = mt * 16 + mrow;
                bf16x8 ah0 = *(const bf16x8*)&Ah[r][ka];
                bf16x8 ah1 = *(const bf16x8*)&Ah[r][ka + 8];
                bf16x8 al0 = *(const bf16x8*)&Al[r][ka];
                bf16x8 al1 = *(const bf16x8*)&Al[r][ka + 8];
                bf16x16 a_hi = __builtin_shufflevector(ah0, ah1,
                    0,1,2,3,4,5,6,7,8,9,10,11,12,13,14,15);
                bf16x16 a_lo = __builtin_shufflevector(al0, al1,
                    0,1,2,3,4,5,6,7,8,9,10,11,12,13,14,15);

                // bf16x3: hi*hi + lo*hi + hi*lo (f32 accumulate)
                acc[mt] = __builtin_amdgcn_wmma_f32_16x16x32_bf16(
                    false, a_hi, false, bh, (short)0, acc[mt], false, false);
                acc[mt] = __builtin_amdgcn_wmma_f32_16x16x32_bf16(
                    false, a_lo, false, bh, (short)0, acc[mt], false, false);
                acc[mt] = __builtin_amdgcn_wmma_f32_16x16x32_bf16(
                    false, a_hi, false, bl, (short)0, acc[mt], false, false);
            }
        }

        // Epilogue: logits -> softplus, reduce 16 columns per row, atomic add.
        const float sb = sbias[n];
#pragma unroll
        for (int mt = 0; mt < 4; ++mt) {
#pragma unroll
            for (int v = 0; v < 8; ++v) {
                float s = softplus_fast(acc[mt][v] + sb);
                s += __shfl_xor(s, 1, 32);
                s += __shfl_xor(s, 2, 32);
                s += __shfl_xor(s, 4, 32);
                s += __shfl_xor(s, 8, 32);
                if ((lane & 15) == 0)
                    atomicAdd(&row_sum[mbase + mt * 16 + v + half * 8], s);
            }
        }
    }
}

// ---------------------------------------------------------------------------
// Kernel 3: true logits (one wave per batch row) -> softplus(-logit).
// ---------------------------------------------------------------------------
__global__ __launch_bounds__(256)
void nce_true(const float* __restrict__ emb, const int* __restrict__ tgt,
              const float* __restrict__ W, const float* __restrict__ bias,
              float* __restrict__ true_part) {
    const int b     = blockIdx.x * 8 + (threadIdx.x >> 5);
    const int lane  = threadIdx.x & 31;
    const int label = tgt[b];

    const float4* e4 = (const float4*)(emb + (size_t)b * EMBED);
    const float4* w4 = (const float4*)(W + (size_t)label * EMBED);
    float dot = 0.0f;
#pragma unroll
    for (int i = lane; i < EMBED / 4; i += 32) {
        float4 a = e4[i], w = w4[i];
        dot = fmaf(a.x, w.x, dot);
        dot = fmaf(a.y, w.y, dot);
        dot = fmaf(a.z, w.z, dot);
        dot = fmaf(a.w, w.w, dot);
    }
    dot += __shfl_xor(dot, 1, 32);
    dot += __shfl_xor(dot, 2, 32);
    dot += __shfl_xor(dot, 4, 32);
    dot += __shfl_xor(dot, 8, 32);
    dot += __shfl_xor(dot, 16, 32);

    if (lane == 0) {
        float logit = dot + bias[label] + neg_log_expected_count(label);
        true_part[b] = softplus_f(-logit);   // sigmoid_xent(logit, 1)
    }
}

// ---------------------------------------------------------------------------
// Kernel 4: mean over BATCH of (true_part + row_sum) -> out[0].
// ---------------------------------------------------------------------------
__global__ __launch_bounds__(256)
void nce_reduce(const float* __restrict__ true_part,
                const float* __restrict__ row_sum, float* __restrict__ out) {
    __shared__ float sm[256];
    float s = 0.0f;
    for (int i = threadIdx.x; i < BATCH; i += 256)
        s += true_part[i] + row_sum[i];
    sm[threadIdx.x] = s;
    __syncthreads();
    for (int off = 128; off > 0; off >>= 1) {
        if (threadIdx.x < off) sm[threadIdx.x] += sm[threadIdx.x + off];
        __syncthreads();
    }
    if (threadIdx.x == 0) out[0] = sm[0] * (1.0f / (float)BATCH);
}

// ---------------------------------------------------------------------------
extern "C" void kernel_launch(void* const* d_in, const int* in_sizes, int n_in,
                              void* d_out, int out_size, void* d_ws, size_t ws_size,
                              hipStream_t stream) {
    (void)in_sizes; (void)n_in; (void)out_size; (void)ws_size;

    const float* emb  = (const float*)d_in[0];   // (BATCH, EMBED) f32
    const int*   tgt  = (const int*)d_in[1];     // (BATCH, 1) i32
    const int*   sids = (const int*)d_in[2];     // (NSAMP,) i32
    const float* W    = (const float*)d_in[3];   // (VOCAB, EMBED) f32
    const float* bias = (const float*)d_in[4];   // (VOCAB,) f32
    float* out = (float*)d_out;

    // Workspace layout (256B-aligned regions).
    char* ws = (char*)d_ws;
    const size_t MB = 1024 * 1024;
    __bf16* wh       = (__bf16*)(ws);                       // 2 MB
    __bf16* wl       = (__bf16*)(ws + 2 * MB);              // 2 MB
    __bf16* eh       = (__bf16*)(ws + 4 * MB);              // 8 MB
    __bf16* el       = (__bf16*)(ws + 12 * MB);             // 8 MB
    float*  sbias    = (float*)(ws + 20 * MB);              // 16 KB
    float*  row_sum  = (float*)(ws + 20 * MB + 65536);      // 64 KB
    float*  true_prt = (float*)(ws + 20 * MB + 131072);     // 64 KB

    hipMemsetAsync(row_sum, 0, BATCH * sizeof(float), stream);

    nce_emb_split<<<(BATCH * EMBED) / 256, 256, 0, stream>>>(emb, eh, el);
    nce_prep<<<NSAMP, 256, 0, stream>>>(sids, W, bias, wh, wl, sbias);
    nce_gemm<<<dim3(8, BATCH / 64), 256, 0, stream>>>(eh, el, wh, wl, sbias, row_sum);
    nce_true<<<BATCH / 8, 256, 0, stream>>>(emb, tgt, W, bias, true_prt);
    nce_reduce<<<1, 256, 0, stream>>>(true_prt, row_sum, out);
}
